// Attention_35296041239224
// MI455X (gfx1250) — matrix operand order repaired and checked by reading, not compiled, not run on hardware
//
#include <hip/hip_runtime.h>
#include <math.h>

// Problem constants (match reference)
#define Bsz 8
#define Hh  8
#define Nn  2048
#define Mm  2048
#define Dd  512

typedef __attribute__((ext_vector_type(16))) __bf16 v16bf;
typedef __attribute__((ext_vector_type(8)))  __bf16 v8bf;
typedef __attribute__((ext_vector_type(8)))  float  v8f;
typedef __attribute__((ext_vector_type(4)))  unsigned int u32x4;
typedef __attribute__((ext_vector_type(4)))  unsigned int v4u;
typedef __attribute__((ext_vector_type(8)))  int v8i;
typedef __attribute__((ext_vector_type(4)))  int v4i;

// Build a 16-element bf16 A/B fragment from two 16-byte chunks.
// Per CDNA5 ISA layouts:
//  A (16x32 bf16): lane row = lane&15; elements e: K=(e>>3)*16 + half*8 + (e&7)
//                  -> chunk0 at K=half*8, chunk1 at K=16+half*8
//  B (32x16 bf16): lane col = lane&15; elements e: K=(e>>3)*16 + half*8 + (e&7)
//                  -> chunk0 at K=half*8, chunk1 at K=16+half*8
__device__ __forceinline__ v16bf frag16(const __bf16* lo, const __bf16* hi) {
  v8bf a = *(const v8bf*)lo;
  v8bf b = *(const v8bf*)hi;
  v16bf r;
#pragma unroll
  for (int e = 0; e < 8; ++e) { r[e] = a[e]; r[e + 8] = b[e]; }
  return r;
}

__device__ __forceinline__ v8f wmma_bf16(v16bf a, v16bf b, v8f c) {
  return __builtin_amdgcn_wmma_f32_16x16x32_bf16(false, a, false, b, (short)0, c,
                                                 false, false);
}

// ---------------------------------------------------------------- TDM
// Issue a 2-D Tensor-Data-Mover load (global -> LDS) per CDNA5 D# layout
// (CDNA5 ISA). data_size = 2 bytes (bf16). Optional LDS
// padding: pad_interval_code/pad_amount_code per D# group1 encodings.
// Tracked by TENSORcnt; caller must s_wait_tensorcnt before consuming.
// This toolchain exposes the 6-arg builtin:
//   (uint32x4 g0, int32x8 g1, int32x4 g2, int32x4 g3, int32x8 gx, i32 cpol)
__device__ __forceinline__ void tdm_load_2d_bf16(
    unsigned lds_byte_off, const void* gptr,
    unsigned tile_dim0_elems,  // tile width in elements (K contiguous)
    unsigned tile_dim1_rows,   // tile height (rows)
    unsigned tensor_dim0, unsigned tensor_dim1,
    unsigned long long dim0_stride_elems,
    int pad_enable, int pad_interval_code, int pad_amount_code) {
  unsigned long long ga = (unsigned long long)gptr;
  v4u g0;
  g0[0] = 1u;                                   // count=1, user mode, no gather
  g0[1] = lds_byte_off;                         // [63:32]  lds_addr (bytes)
  g0[2] = (unsigned)(ga & 0xffffffffu);         // [95:64]  global_addr lo
  g0[3] = (unsigned)((ga >> 32) & 0x01ffffffu)  // [120:96] global_addr hi
          | (2u << 30);                         // [127:126] type = 2 (image)

  v8i g1;
  g1[0] = (int)((1u << 16)                              // data_size: 2 bytes
                | ((unsigned)pad_enable << 20)
                | ((unsigned)pad_interval_code << 22)
                | ((unsigned)pad_amount_code << 25));   // wg_mask=0 (no cluster)
  g1[1] = (int)((tensor_dim0 & 0xffffu) << 16);         // [63:48] dim0[15:0]
  g1[2] = (int)(((tensor_dim0 >> 16) & 0xffffu) |
                ((tensor_dim1 & 0xffffu) << 16));       // dim0 hi | dim1 lo
  g1[3] = (int)(((tensor_dim1 >> 16) & 0xffffu) |
                ((tile_dim0_elems & 0xffffu) << 16));   // dim1 hi | tile_dim0
  g1[4] = (int)(tile_dim1_rows & 0xffffu);              // tile_dim1, tile_dim2=0
  g1[5] = (int)(unsigned)(dim0_stride_elems & 0xffffffffu);
  g1[6] = (int)(unsigned)((dim0_stride_elems >> 32) & 0xffffu);  // stride hi
  g1[7] = 0;
  v4i gz4 = {0, 0, 0, 0};        // 2-D tensor: groups 2/3 unused
  v8i gz8 = {0, 0, 0, 0, 0, 0, 0, 0};
  __builtin_amdgcn_tensor_load_to_lds(g0, g1, gz4, gz4, gz8, 0);
}

// ---------------------------------------------------------------- convert
__global__ void cvt_f32_bf16(const float* __restrict__ in,
                             __bf16* __restrict__ out, long n) {
  long i = (long)blockIdx.x * blockDim.x + threadIdx.x;
  long stride = (long)gridDim.x * blockDim.x;
  for (; i < n; i += stride) out[i] = (__bf16)in[i];
}

// ---------------------------------------------------------------- GEMM
// C[z] = (A[z/Hdiv] @ W[z%Hdiv] + bias[z%Hdiv]) * scale
// A: [Mrows,K] bf16 row-major, W: [K,Ncols] bf16 row-major.
// Block tile 128x128, K-step 32, 256 threads = 8 waves (4 along M x 2 along N),
// wave tile 32x64 = 2x4 WMMA tiles. The A tile is staged by the Tensor Data
// Mover (TDM) with hardware LDS padding (16-DWORD interval + 4-DWORD pad ->
// 80B row pitch = G_LS bf16), so SIMDs never touch the A copy.
#define G_BM 128
#define G_BN 128
#define G_BK 32
#define G_LS 40  // LDS row stride in bf16 (80B: 16B aligned, bank-spread)

__global__ void __launch_bounds__(256)
gemm_bf16_kernel(const __bf16* __restrict__ A, const __bf16* __restrict__ W,
                 const float* __restrict__ bias,
                 __bf16* __restrict__ Cb, float* __restrict__ Cf,
                 int Mrows, int Ncols, int K,
                 long long strideA, long long strideW, long long strideC,
                 int Hdiv, float scale, int transposeC) {
  __shared__ __bf16 As[G_BM][G_LS];   // A tile, K contiguous per row (TDM-fed)
  __shared__ __bf16 Wt[G_BN][G_LS];   // W^T tile: [n][k], K contiguous per row

  const int tid  = threadIdx.x;
  const int lane = tid & 31;
  const int wave = tid >> 5;
  const int half = lane >> 4;
  const int l16  = lane & 15;
  const int z  = blockIdx.z;
  const int n0 = blockIdx.x * G_BN;
  const int m0 = blockIdx.y * G_BM;

  const __bf16* Ab = A + (size_t)(z / Hdiv) * strideA;
  const __bf16* Wb = W + (size_t)(z % Hdiv) * strideW;
  const float* biasb = bias + (size_t)(z % Hdiv) * Ncols;

  // LDS byte offset of As: flat->LDS mapping truncates to addr[31:0].
  const unsigned as_off = (unsigned)(size_t)(&As[0][0]);

  const int wm = wave & 3;   // 0..3 along M (rows wm*32)
  const int wn = wave >> 2;  // 0..1 along N (cols wn*64)

  v8f czero = {};
  v8f c[2][4];
#pragma unroll
  for (int i = 0; i < 2; ++i)
#pragma unroll
    for (int j = 0; j < 4; ++j) c[i][j] = czero;

  for (int kt = 0; kt < K; kt += G_BK) {
    __syncthreads();
    // Stage A tile 128x32 via TDM: tile_dim0=32 elems (64B rows, padded to
    // 80B in LDS), tile_dim1=128 rows, row stride = K elements.
    if (wave == 0) {
      tdm_load_2d_bf16(as_off, Ab + (size_t)m0 * K + kt,
                       G_BK, G_BM, (unsigned)K, (unsigned)Mrows,
                       (unsigned long long)K,
                       /*pad_enable=*/1, /*interval 16 DW=*/3, /*pad 4 DW=*/3);
    }
    // Stage W^T tile: read row-major W (k, 8 contiguous n), scatter to [n][k].
#pragma unroll
    for (int c0 = 0; c0 < 2; ++c0) {
      int chunk = tid + c0 * 256;
      int krow = chunk >> 4;
      int nc   = (chunk & 15) * 8;
      const __bf16* wp = Wb + (size_t)(kt + krow) * Ncols + n0 + nc;
      v8bf wv = *(const v8bf*)wp;
      if (kt + G_BK < K) __builtin_prefetch(wp + (size_t)G_BK * Ncols, 0, 0);
#pragma unroll
      for (int j = 0; j < 8; ++j) Wt[nc + j][krow] = wv[j];
    }
    if (wave == 0) __builtin_amdgcn_s_wait_tensorcnt(0);
    __syncthreads();

    v16bf a[2], bw[4];
#pragma unroll
    for (int i = 0; i < 2; ++i) {
      int row = wm * 32 + i * 16 + l16;
      a[i] = frag16(&As[row][half * 8], &As[row][16 + half * 8]);
    }
#pragma unroll
    for (int j = 0; j < 4; ++j) {
      int col = wn * 64 + j * 16 + l16;
      bw[j] = frag16(&Wt[col][half * 8], &Wt[col][16 + half * 8]);
    }
#pragma unroll
    for (int i = 0; i < 2; ++i)
#pragma unroll
      for (int j = 0; j < 4; ++j) c[i][j] = wmma_bf16(a[i], bw[j], c[i][j]);
  }

  // Epilogue: C layout per tile: VGPR r -> row r + 8*half, lane -> col.
#pragma unroll
  for (int i = 0; i < 2; ++i) {
#pragma unroll
    for (int j = 0; j < 4; ++j) {
      int gcol = n0 + wn * 64 + j * 16 + l16;
      float bv = biasb[gcol];
#pragma unroll
      for (int r = 0; r < 8; ++r) {
        int grow = m0 + wm * 32 + i * 16 + r + half * 8;
        float v = (c[i][j][r] + bv) * scale;
        if (Cf) {
          Cf[(size_t)z * strideC + (size_t)grow * Ncols + gcol] = v;
        } else if (transposeC) {
          Cb[(size_t)z * strideC + (size_t)gcol * Mrows + grow] = (__bf16)v;
        } else {
          Cb[(size_t)z * strideC + (size_t)grow * Ncols + gcol] = (__bf16)v;
        }
      }
    }
  }
}

// ---------------------------------------------------------------- flash attention
// qh: [B,H,M,D] bf16 (already scaled by 1/sqrt(D)); kh: [B,H,N,D] bf16;
// vt: [B,H,D,N] bf16 (V transposed); out: [B,M,D*H] bf16, index d*H + h.
// Block: 128 threads = 4 waves, handles 16 q-rows; each wave owns a 128-wide
// D-slice of the output accumulator; QK^T K-reduction split across waves and
// summed through LDS; online softmax; PV via WMMA with K=32 tiles.
#define FA_BM 16
#define FA_BN 32
#define FA_QS 520  // q LDS stride (bf16)
#define FA_PS 40   // P LDS stride (bf16)

__global__ void __launch_bounds__(128)
flash_attn_kernel(const __bf16* __restrict__ qh, const __bf16* __restrict__ kh,
                  const __bf16* __restrict__ vt, __bf16* __restrict__ out) {
  __shared__ __bf16 qs[FA_BM][FA_QS];        // 16x520x2B = 16,640B
  __shared__ float  sred[4][FA_BM * FA_BN];  // 8,192B
  __shared__ __bf16 ps[FA_BM][FA_PS];        // 1,280B

  const int tid  = threadIdx.x;
  const int lane = tid & 31;
  const int wave = tid >> 5;
  const int half = lane >> 4;
  const int l16  = lane & 15;
  const int m0 = blockIdx.x * FA_BM;
  const int h  = blockIdx.y;
  const int b  = blockIdx.z;
  const int d0 = wave * 128;  // this wave's D-slice

  const size_t bh = (size_t)b * Hh + h;
  const __bf16* qg = qh + (bh * Nn + m0) * Dd;
  const __bf16* kg = kh + bh * Nn * Dd;
  const __bf16* vg = vt + bh * (size_t)Dd * Nn;

  // Stage q tile [16][512]: 1024 chunks of 8 bf16, 8 per thread.
#pragma unroll
  for (int c0 = 0; c0 < 8; ++c0) {
    int chunk = tid + c0 * 128;
    int row = chunk >> 6;
    int col = (chunk & 63) * 8;
    *(u32x4*)&qs[row][col] = *(const u32x4*)(qg + (size_t)row * Dd + col);
  }
  __syncthreads();

  v8f czero = {};
  v8f acc[8];
  float mrow[8], lrow[8];
#pragma unroll
  for (int r = 0; r < 8; ++r) { mrow[r] = -1e30f; lrow[r] = 0.f; }
#pragma unroll
  for (int t = 0; t < 8; ++t) acc[t] = czero;

  for (int n0 = 0; n0 < Nn; n0 += FA_BN) {
    // Partial S = q[:, d0:d0+128] @ k[:, d0:d0+128]^T  (two 16x16 tiles)
    v8f s0 = {}, s1 = {};
    const __bf16* krow0 = kg + (size_t)(n0 + l16) * Dd;       // B col = l16
    const __bf16* krow1 = kg + (size_t)(n0 + 16 + l16) * Dd;  // B col = 16+l16
#pragma unroll
    for (int kk = 0; kk < 4; ++kk) {
      int kb = d0 + kk * 32;
      v16bf a  = frag16(&qs[l16][kb + half * 8], &qs[l16][kb + 16 + half * 8]);
      v16bf b0 = frag16(krow0 + kb + half * 8, krow0 + kb + 16 + half * 8);
      v16bf b1 = frag16(krow1 + kb + half * 8, krow1 + kb + 16 + half * 8);
      s0 = wmma_bf16(a, b0, s0);
      s1 = wmma_bf16(a, b1, s1);
    }
    __syncthreads();  // prior iteration's sred/ps reads complete
#pragma unroll
    for (int r = 0; r < 8; ++r) {
      sred[wave][r * 64 + lane]      = s0[r];
      sred[wave][r * 64 + 32 + lane] = s1[r];
    }
    __syncthreads();

    // Full S = sum of wave partials; online softmax (rows r+8*half per lane).
    float alpha[8];
#pragma unroll
    for (int r = 0; r < 8; ++r) {
      float f0 = sred[0][r * 64 + lane] + sred[1][r * 64 + lane] +
                 sred[2][r * 64 + lane] + sred[3][r * 64 + lane];
      float f1 = sred[0][r * 64 + 32 + lane] + sred[1][r * 64 + 32 + lane] +
                 sred[2][r * 64 + 32 + lane] + sred[3][r * 64 + 32 + lane];
      float mx = fmaxf(f0, f1);
#pragma unroll
      for (int off = 1; off <= 8; off <<= 1)
        mx = fmaxf(mx, __shfl_xor(mx, off, 32));
      float mnew = fmaxf(mrow[r], mx);
      float al = __expf(mrow[r] - mnew);
      float p0 = __expf(f0 - mnew);
      float p1 = __expf(f1 - mnew);
      float rs = p0 + p1;
#pragma unroll
      for (int off = 1; off <= 8; off <<= 1) rs += __shfl_xor(rs, off, 32);
      lrow[r] = lrow[r] * al + rs;
      mrow[r] = mnew;
      alpha[r] = al;
      if (wave == 0) {  // all waves compute identical P; one writer
        ps[r + half * 8][l16]      = (__bf16)p0;
        ps[r + half * 8][16 + l16] = (__bf16)p1;
      }
    }
#pragma unroll
    for (int t = 0; t < 8; ++t)
#pragma unroll
      for (int r = 0; r < 8; ++r) acc[t][r] *= alpha[r];
    __syncthreads();  // P visible

    // O[:, d0:d0+128] += P @ V[n0:n0+32, d0:d0+128]   (8 tiles, K=32)
    v16bf pa = frag16(&ps[l16][half * 8], &ps[l16][16 + half * 8]);
#pragma unroll
    for (int t = 0; t < 8; ++t) {
      const __bf16* vrow = vg + (size_t)(d0 + t * 16 + l16) * Nn + n0;
      v16bf bv = frag16(vrow + half * 8, vrow + 16 + half * 8);
      acc[t] = wmma_bf16(pa, bv, acc[t]);
    }
  }

  // Normalize and store: out[b, m, d*H + h]
#pragma unroll
  for (int r = 0; r < 8; ++r) {
    float inv = 1.0f / lrow[r];
    int row = m0 + r + half * 8;
#pragma unroll
    for (int t = 0; t < 8; ++t) {
      int d = d0 + t * 16 + l16;
      float v = acc[t][r] * inv;
      out[((size_t)b * Mm + row) * (Dd * Hh) + (size_t)d * Hh + h] = (__bf16)v;
    }
  }
}

// ---------------------------------------------------------------- launch
extern "C" void kernel_launch(void* const* d_in, const int* in_sizes, int n_in,
                              void* d_out, int out_size, void* d_ws,
                              size_t ws_size, hipStream_t stream) {
  const float* kf = (const float*)d_in[0];
  const float* vf = (const float*)d_in[1];
  const float* qf = (const float*)d_in[2];
  const float* Wk = (const float*)d_in[3];
  const float* bk = (const float*)d_in[4];
  const float* Wv = (const float*)d_in[5];
  const float* bv = (const float*)d_in[6];
  const float* Wq = (const float*)d_in[7];
  const float* bq = (const float*)d_in[8];
  const float* Wo = (const float*)d_in[9];
  const float* bo = (const float*)d_in[10];
  float* outp = (float*)d_out;

  const size_t S_BND  = (size_t)Bsz * Nn * Dd;        // 8,388,608
  const size_t S_W    = (size_t)Hh * Dd * Dd;         // 2,097,152 (Wo same)
  const size_t S_BHND = (size_t)Bsz * Hh * Nn * Dd;   // 67,108,864
  const size_t S_ATT  = (size_t)Bsz * Mm * Dd * Hh;   // 67,108,864

  char* w = (char*)d_ws;
  auto alloc = [&](size_t bytes) {
    char* p = w;
    w += (bytes + 255) & ~(size_t)255;
    return p;
  };
  __bf16* kb  = (__bf16*)alloc(S_BND * 2);
  __bf16* vb  = (__bf16*)alloc(S_BND * 2);
  __bf16* qb  = (__bf16*)alloc(S_BND * 2);
  __bf16* wkb = (__bf16*)alloc(S_W * 2);
  __bf16* wvb = (__bf16*)alloc(S_W * 2);
  __bf16* wqb = (__bf16*)alloc(S_W * 2);
  __bf16* wob = (__bf16*)alloc(S_W * 2);
  __bf16* qhp = (__bf16*)alloc(S_BHND * 2);  // [B,H,M,D]
  __bf16* khp = (__bf16*)alloc(S_BHND * 2);  // [B,H,N,D]
  __bf16* vtp = (__bf16*)alloc(S_BHND * 2);  // [B,H,D,N]
  __bf16* att = (__bf16*)alloc(S_ATT * 2);   // [B,M,D*H]
  // Requires ws_size >= ~576 MiB.

  // 1) f32 -> bf16 conversions
  cvt_f32_bf16<<<1024, 256, 0, stream>>>(kf, kb, (long)S_BND);
  cvt_f32_bf16<<<1024, 256, 0, stream>>>(vf, vb, (long)S_BND);
  cvt_f32_bf16<<<1024, 256, 0, stream>>>(qf, qb, (long)S_BND);
  cvt_f32_bf16<<<1024, 256, 0, stream>>>(Wk, wkb, (long)S_W);
  cvt_f32_bf16<<<1024, 256, 0, stream>>>(Wv, wvb, (long)S_W);
  cvt_f32_bf16<<<1024, 256, 0, stream>>>(Wq, wqb, (long)S_W);
  cvt_f32_bf16<<<1024, 256, 0, stream>>>(Wo, wob, (long)S_W);

  // 2) Per-head projections: grid (N/128, M/128, B*H)
  dim3 pgrid(Dd / G_BN, Nn / G_BM, Bsz * Hh);
  const long long sA = (long long)Nn * Dd;
  const long long sW = (long long)Dd * Dd;
  const long long sC = (long long)Nn * Dd;
  const float qscale = 0.04419417382415922f;  // 1/sqrt(512): fold score scale into q_
  gemm_bf16_kernel<<<pgrid, 256, 0, stream>>>(qb, wqb, bq, qhp, nullptr, Nn, Dd,
                                              Dd, sA, sW, sC, Hh, qscale, 0);
  gemm_bf16_kernel<<<pgrid, 256, 0, stream>>>(kb, wkb, bk, khp, nullptr, Nn, Dd,
                                              Dd, sA, sW, sC, Hh, 1.0f, 0);
  gemm_bf16_kernel<<<pgrid, 256, 0, stream>>>(vb, wvb, bv, vtp, nullptr, Nn, Dd,
                                              Dd, sA, sW, sC, Hh, 1.0f, 1);

  // 3) Flash attention: grid (M/16, H, B), 128 threads
  flash_attn_kernel<<<dim3(Mm / FA_BM, Hh, Bsz), 128, 0, stream>>>(qhp, khp,
                                                                   vtp, att);

  // 4) Final projection: [B*M, H*D] @ [H*D, D] + bo -> f32 d_out
  dim3 fgrid(Dd / G_BN, (Bsz * Mm) / G_BM, 1);
  gemm_bf16_kernel<<<fgrid, 256, 0, stream>>>(att, wob, bo, nullptr, outp,
                                              Bsz * Mm, Dd, (long long)Hh * Dd,
                                              0, 0, 0, 1, 1.0f, 0);
}
